// QuantizerWrapper_62277025792422
// MI455X (gfx1250) — matrix operand order repaired and checked by reading, not compiled, not run on hardware
//
#include <hip/hip_runtime.h>
#include <hip/hip_bf16.h>

typedef __bf16 bf16_t;
typedef bf16_t v16bf __attribute__((ext_vector_type(16)));
typedef bf16_t v8bf  __attribute__((ext_vector_type(8)));
typedef float  v8f   __attribute__((ext_vector_type(8)));
typedef int    v4i_  __attribute__((ext_vector_type(4)));

#define B_   16
#define D_   512
#define T_   2000
#define Q_   8
#define C_   1024
#define M_   (B_ * T_)          /* 32000 rows */

#define MT    64                 /* rows per workgroup             */
#define NTI   256                /* cols per N-iteration (4 iters) */
#define KC    32                 /* K chunk (one WMMA k-step)      */
#define NK    (D_ / KC)          /* 16 chunks                      */
#define PITCH 40                 /* KC + 8; 80B rows: 16B-aligned, conflict-free */

// ---------------------------------------------------------------------------
// gfx1250 async global->LDS path (fallback: synchronous uint4 copy)
#if defined(__gfx1250__) && __has_builtin(__builtin_amdgcn_global_load_async_to_lds_b128) && \
    __has_builtin(__builtin_amdgcn_s_wait_asynccnt)
#define USE_ASYNC 1
#else
#define USE_ASYNC 0
#endif

__device__ __forceinline__ void cp16(const bf16_t* g, bf16_t* l) {
#if USE_ASYNC
  __builtin_amdgcn_global_load_async_to_lds_b128(
      (__attribute__((address_space(1))) v4i_*)g,
      (__attribute__((address_space(3))) v4i_*)l, 0, 0);
#else
  *(uint4*)l = *(const uint4*)g;
#endif
}

#if USE_ASYNC
#define WAIT_ASYNC(n) __builtin_amdgcn_s_wait_asynccnt(n)
#else
#define WAIT_ASYNC(n) ((void)0)
#endif

// ---------------------------------------------------------------------------
// Fragment loaders per CDNA5 ISA VGPR layouts (CDNA5 ISA §7.12.2), KC==32
// A 16x32 bf16: lane half h holds K = {8h..8h+7} U {16+8h..23+8h}
__device__ __forceinline__ v16bf frag_a(const bf16_t* rowp, int h) {
  v8bf a = *(const v8bf*)(rowp + 8 * h);
  v8bf b = *(const v8bf*)(rowp + 16 + 8 * h);
  v16bf f;
#pragma unroll
  for (int j = 0; j < 8; ++j) { f[j] = a[j]; f[j + 8] = b[j]; }
  return f;
}
// B 32x16 bf16 (stored N-major = codebook rows): lane half h holds K = {8h..8h+7} U {16+8h..23+8h}
__device__ __forceinline__ v16bf frag_b(const bf16_t* rowp, int h) {
  v8bf a = *(const v8bf*)(rowp + 8 * h);
  v8bf b = *(const v8bf*)(rowp + 16 + 8 * h);
  v16bf f;
#pragma unroll
  for (int j = 0; j < 8; ++j) { f[j] = a[j]; f[j + 8] = b[j]; }
  return f;
}

// ---------------------------------------------------------------------------
// x[B,D,T] -> residual[B*T, D] (f32 + bf16 copies)
__global__ __launch_bounds__(256) void k_transpose_in(const float* __restrict__ x,
                                                      float* __restrict__ res,
                                                      bf16_t* __restrict__ resb) {
  __shared__ float tile[32][33];
  const int b = blockIdx.z, t0 = blockIdx.x * 32, d0 = blockIdx.y * 32;
  const int tx = threadIdx.x, ty = threadIdx.y;
#pragma unroll
  for (int i = 0; i < 4; ++i) {
    int d = d0 + ty + 8 * i, t = t0 + tx;
    if (t < T_) tile[ty + 8 * i][tx] = x[((size_t)b * D_ + d) * T_ + t];
  }
  __syncthreads();
#pragma unroll
  for (int i = 0; i < 4; ++i) {
    int t = t0 + ty + 8 * i, d = d0 + tx;
    if (t < T_) {
      float v = tile[tx][ty + 8 * i];
      size_t o = ((size_t)b * T_ + t) * D_ + d;
      res[o]  = v;
      resb[o] = (bf16_t)v;
    }
  }
}

// codebooks f32 -> bf16 + ||c||^2; one wave per (q, code) row
__global__ __launch_bounds__(256) void k_prep_cb(const float* __restrict__ cb,
                                                 bf16_t* __restrict__ cbb,
                                                 float* __restrict__ cnorm) {
  const int lane = threadIdx.x & 31;
  const int wid = (blockIdx.x * blockDim.x + threadIdx.x) >> 5;  // 0..Q_*C_-1
  const float* src = cb + (size_t)wid * D_;
  bf16_t* dst = cbb + (size_t)wid * D_;
  float s = 0.f;
#pragma unroll
  for (int i = 0; i < 2; ++i) {
    int e = i * 256 + lane * 8;
    float4 v0 = *(const float4*)(src + e);
    float4 v1 = *(const float4*)(src + e + 4);
    union { bf16_t h[8]; uint4 u; } p;
    p.h[0] = (bf16_t)v0.x; p.h[1] = (bf16_t)v0.y; p.h[2] = (bf16_t)v0.z; p.h[3] = (bf16_t)v0.w;
    p.h[4] = (bf16_t)v1.x; p.h[5] = (bf16_t)v1.y; p.h[6] = (bf16_t)v1.z; p.h[7] = (bf16_t)v1.w;
    *(uint4*)(dst + e) = p.u;
    s += v0.x * v0.x + v0.y * v0.y + v0.z * v0.z + v0.w * v0.w;
    s += v1.x * v1.x + v1.y * v1.y + v1.z * v1.z + v1.w * v1.w;
  }
#pragma unroll
  for (int off = 16; off > 0; off >>= 1) s += __shfl_xor(s, off, 32);
  if (lane == 0) cnorm[wid] = s;
}

// ---------------------------------------------------------------------------
// WMMA GEMM + per-row argmin of (||c||^2 - 2 * <r, c>)
// 8 waves as 2(wm) x 4(wn); each wave computes 32 rows x 64 cols.
// 3 LDS buffers, 2 async chunks in flight, 1 barrier per K-chunk.
// K and N loops kept ROLLED to bound register pressure (no spills).
__global__ __launch_bounds__(256) void k_gemm_argmin(const bf16_t* __restrict__ resb,
                                                     const bf16_t* __restrict__ cbb,
                                                     const float* __restrict__ cnorm,
                                                     int* __restrict__ amin) {
  __shared__ bf16_t sA[3][MT * PITCH];
  __shared__ bf16_t sB[3][NTI * PITCH];
  __shared__ float  sBv[MT][4];
  __shared__ int    sBi[MT][4];

  const int tid  = threadIdx.x;
  const int lane = tid & 31, w = tid >> 5;
  const int wm = w & 1, wn = w >> 1;        // 2 x 4 wave grid
  const int h = lane >> 4, ln = lane & 15;
  const int m_base = blockIdx.x * MT;

  const int arow = tid >> 2, aseg = (tid & 3) * 8;   // A: 1 x b128 per thread

  float bestv[2][8];
  int   besti[2][8];
#pragma unroll
  for (int s = 0; s < 2; ++s)
#pragma unroll
    for (int r = 0; r < 8; ++r) { bestv[s][r] = 3.4e38f; besti[s][r] = 0; }

#pragma unroll 1
  for (int n_base = 0; n_base < C_; n_base += NTI) {
    auto copy_chunk = [&](int kb, int buf) {
      cp16(resb + (size_t)(m_base + arow) * D_ + kb * KC + aseg,
           &sA[buf][arow * PITCH + aseg]);
#pragma unroll
      for (int i = 0; i < 4; ++i) {
        int lin = tid + 256 * i;               // 0..1023
        int brow = lin >> 2, bseg = (lin & 3) * 8;
        cp16(cbb + (size_t)(n_base + brow) * D_ + kb * KC + bseg,
             &sB[buf][brow * PITCH + bseg]);
      }
    };

    // hoist cnorm for this N-slab (consumed only in the epilogue)
    float cn[4];
#pragma unroll
    for (int t = 0; t < 4; ++t) cn[t] = cnorm[n_base + wn * 64 + t * 16 + ln];

    v8f acc[8];                                // [s*4 + t]
#pragma unroll
    for (int a = 0; a < 8; ++a)
#pragma unroll
      for (int j = 0; j < 8; ++j) acc[a][j] = 0.f;

    // all fragment loads first -> progressive dscnt waits, dense WMMA issue
    auto compute = [&](int buf) {
      const bf16_t* pA = sA[buf];
      const bf16_t* pB = sB[buf];
      v16bf af0 = frag_a(pA + (wm * 32 + ln) * PITCH, h);
      v16bf af1 = frag_a(pA + (wm * 32 + 16 + ln) * PITCH, h);
      v16bf bf0 = frag_b(pB + (wn * 64 + 0 * 16 + ln) * PITCH, h);
      v16bf bf1 = frag_b(pB + (wn * 64 + 1 * 16 + ln) * PITCH, h);
      v16bf bf2 = frag_b(pB + (wn * 64 + 2 * 16 + ln) * PITCH, h);
      v16bf bf3 = frag_b(pB + (wn * 64 + 3 * 16 + ln) * PITCH, h);
      acc[0] = __builtin_amdgcn_wmma_f32_16x16x32_bf16(false, af0, false, bf0, (short)0, acc[0], false, false);
      acc[4] = __builtin_amdgcn_wmma_f32_16x16x32_bf16(false, af1, false, bf0, (short)0, acc[4], false, false);
      acc[1] = __builtin_amdgcn_wmma_f32_16x16x32_bf16(false, af0, false, bf1, (short)0, acc[1], false, false);
      acc[5] = __builtin_amdgcn_wmma_f32_16x16x32_bf16(false, af1, false, bf1, (short)0, acc[5], false, false);
      acc[2] = __builtin_amdgcn_wmma_f32_16x16x32_bf16(false, af0, false, bf2, (short)0, acc[2], false, false);
      acc[6] = __builtin_amdgcn_wmma_f32_16x16x32_bf16(false, af1, false, bf2, (short)0, acc[6], false, false);
      acc[3] = __builtin_amdgcn_wmma_f32_16x16x32_bf16(false, af0, false, bf3, (short)0, acc[3], false, false);
      acc[7] = __builtin_amdgcn_wmma_f32_16x16x32_bf16(false, af1, false, bf3, (short)0, acc[7], false, false);
    };

    copy_chunk(0, 0);                          // prologue: 2 chunks in flight
    copy_chunk(1, 1);
    int cur = 0, pre = 2;                      // pre == (kb+2)%3
#pragma unroll 1
    for (int kb = 0; kb < NK - 1; ++kb) {      // steady state: unconditional wait(5)
      WAIT_ASYNC(5);                           // in-order: chunk kb landed
      __syncthreads();                         // chunk kb visible; kb-1 compute done
      if (kb + 2 < NK) copy_chunk(kb + 2, pre);
      compute(cur);
      cur = (cur == 2) ? 0 : cur + 1;
      pre = (pre == 2) ? 0 : pre + 1;
    }
    WAIT_ASYNC(0);                             // drain last chunk
    __syncthreads();
    compute(cur);
    __syncthreads();                           // last chunk consumed before reuse

    // score = cnorm[n] - 2*dot ; running argmin
#pragma unroll
    for (int t = 0; t < 4; ++t) {
      int n = n_base + wn * 64 + t * 16 + ln;
#pragma unroll
      for (int r = 0; r < 8; ++r) {
        float s0 = cn[t] - 2.f * acc[t][r];
        if (s0 < bestv[0][r]) { bestv[0][r] = s0; besti[0][r] = n; }
        float s1 = cn[t] - 2.f * acc[4 + t][r];
        if (s1 < bestv[1][r]) { bestv[1][r] = s1; besti[1][r] = n; }
      }
    }
  }

  // reduce over the 16 lanes holding row (s,r): m = wm*32 + s*16 + r + 8h
#pragma unroll
  for (int s = 0; s < 2; ++s)
#pragma unroll
    for (int r = 0; r < 8; ++r) {
      float v = bestv[s][r];
      int   i = besti[s][r];
#pragma unroll
      for (int off = 1; off < 16; off <<= 1) {
        float ov = __shfl_xor(v, off, 32);
        int   oi = __shfl_xor(i, off, 32);
        if (ov < v || (ov == v && oi < i)) { v = ov; i = oi; }
      }
      if (ln == 0) {
        int rl = wm * 32 + s * 16 + r + 8 * h;
        sBv[rl][wn] = v; sBi[rl][wn] = i;
      }
    }
  __syncthreads();
  if (tid < MT) {
    float bv = sBv[tid][0];
    int   bi = sBi[tid][0];
#pragma unroll
    for (int j = 1; j < 4; ++j) {
      float v = sBv[tid][j]; int i = sBi[tid][j];
      if (v < bv || (v == bv && i < bi)) { bv = v; bi = i; }
    }
    amin[m_base + tid] = bi;
  }
}

// ---------------------------------------------------------------------------
// gather code; residual -= qv (f32 + bf16); qacc (+)= qv; per-row sum(nr^2)
__global__ __launch_bounds__(256) void k_update(float* __restrict__ res,
                                                bf16_t* __restrict__ resb,
                                                const float* __restrict__ cb,
                                                const int* __restrict__ amin,
                                                float* __restrict__ qacc,
                                                float* __restrict__ loss_rows,
                                                int init) {
  const int lane = threadIdx.x & 31;
  const int wv   = threadIdx.x >> 5;
  const int row  = blockIdx.x * 8 + wv;
  const int idx  = amin[row];
  const float* cr = cb + (size_t)idx * D_;
  float* rr = res + (size_t)row * D_;
  float* qr = qacc + (size_t)row * D_;
  bf16_t* rb = resb + (size_t)row * D_;
  float ss = 0.f;
#pragma unroll
  for (int i = 0; i < 2; ++i) {
    int e = i * 256 + lane * 8;
    float4 q0 = *(const float4*)(cr + e);
    float4 q1 = *(const float4*)(cr + e + 4);
    float4 r0 = *(const float4*)(rr + e);
    float4 r1 = *(const float4*)(rr + e + 4);
    float4 n0 = make_float4(r0.x - q0.x, r0.y - q0.y, r0.z - q0.z, r0.w - q0.w);
    float4 n1 = make_float4(r1.x - q1.x, r1.y - q1.y, r1.z - q1.z, r1.w - q1.w);
    *(float4*)(rr + e)     = n0;
    *(float4*)(rr + e + 4) = n1;
    union { bf16_t hh[8]; uint4 u; } p;
    p.hh[0] = (bf16_t)n0.x; p.hh[1] = (bf16_t)n0.y; p.hh[2] = (bf16_t)n0.z; p.hh[3] = (bf16_t)n0.w;
    p.hh[4] = (bf16_t)n1.x; p.hh[5] = (bf16_t)n1.y; p.hh[6] = (bf16_t)n1.z; p.hh[7] = (bf16_t)n1.w;
    *(uint4*)(rb + e) = p.u;
    ss += n0.x * n0.x + n0.y * n0.y + n0.z * n0.z + n0.w * n0.w;
    ss += n1.x * n1.x + n1.y * n1.y + n1.z * n1.z + n1.w * n1.w;
    if (init) {
      *(float4*)(qr + e) = q0;
      *(float4*)(qr + e + 4) = q1;
    } else {
      float4 p0 = *(const float4*)(qr + e);
      float4 p1 = *(const float4*)(qr + e + 4);
      *(float4*)(qr + e)     = make_float4(p0.x + q0.x, p0.y + q0.y, p0.z + q0.z, p0.w + q0.w);
      *(float4*)(qr + e + 4) = make_float4(p1.x + q1.x, p1.y + q1.y, p1.z + q1.z, p1.w + q1.w);
    }
  }
#pragma unroll
  for (int off = 16; off > 0; off >>= 1) ss += __shfl_xor(ss, off, 32);
  if (lane == 0) {
    if (init) loss_rows[row] = ss;
    else      loss_rows[row] += ss;
  }
}

// deterministic loss reduction (single block, fixed order)
__global__ __launch_bounds__(256) void k_reduce_loss(const float* __restrict__ loss_rows,
                                                     float* __restrict__ loss) {
  __shared__ float s[256];
  const int tid = threadIdx.x;
  float acc = 0.f;
  for (int i = tid; i < M_; i += 256) acc += loss_rows[i];
  s[tid] = acc;
  __syncthreads();
  for (int st = 128; st > 0; st >>= 1) {
    if (tid < st) s[tid] += s[tid + st];
    __syncthreads();
  }
  if (tid == 0) loss[0] = s[0];
}

// qacc[B*T, D] -> out[B,D,T] ; append scalar loss
__global__ __launch_bounds__(256) void k_finalize(const float* __restrict__ qacc,
                                                  float* __restrict__ out,
                                                  const float* __restrict__ loss) {
  __shared__ float tile[32][33];
  const int b = blockIdx.z, t0 = blockIdx.x * 32, d0 = blockIdx.y * 32;
  const int tx = threadIdx.x, ty = threadIdx.y;
#pragma unroll
  for (int i = 0; i < 4; ++i) {
    int t = t0 + ty + 8 * i;
    if (t < T_) tile[ty + 8 * i][tx] = qacc[((size_t)b * T_ + t) * D_ + d0 + tx];
  }
  __syncthreads();
#pragma unroll
  for (int i = 0; i < 4; ++i) {
    int t = t0 + tx, d = d0 + ty + 8 * i;
    if (t < T_) out[((size_t)b * D_ + d) * T_ + t] = tile[tx][ty + 8 * i];
  }
  if (blockIdx.x == 0 && blockIdx.y == 0 && blockIdx.z == 0 && tx == 0 && ty == 0)
    out[(size_t)B_ * D_ * T_] = loss[0] * (1.0f / (float)((size_t)B_ * T_ * D_));
}

// ---------------------------------------------------------------------------
extern "C" void kernel_launch(void* const* d_in, const int* in_sizes, int n_in,
                              void* d_out, int out_size, void* d_ws, size_t ws_size,
                              hipStream_t stream) {
  (void)in_sizes; (void)n_in; (void)out_size; (void)ws_size;
  const float* x  = (const float*)d_in[0];   // [B, D, T]
  const float* cb = (const float*)d_in[1];   // [Q, C, D]
  float* out = (float*)d_out;                // [B, D, T] + 1 scalar

  char* w = (char*)d_ws;
  float*  residual  = (float*)w;  w += (size_t)M_ * D_ * 4;
  float*  qacc      = (float*)w;  w += (size_t)M_ * D_ * 4;
  bf16_t* res_bf    = (bf16_t*)w; w += (size_t)M_ * D_ * 2;
  bf16_t* cb_bf     = (bf16_t*)w; w += (size_t)Q_ * C_ * D_ * 2;
  float*  cnorm     = (float*)w;  w += (size_t)Q_ * C_ * 4;
  float*  loss_rows = (float*)w;  w += (size_t)M_ * 4;
  float*  loss      = (float*)w;  w += 64;
  int*    amin      = (int*)w;

  dim3 tb(32, 8);
  dim3 tg((T_ + 31) / 32, D_ / 32, B_);
  k_transpose_in<<<tg, tb, 0, stream>>>(x, residual, res_bf);
  k_prep_cb<<<(Q_ * C_ * 32) / 256, 256, 0, stream>>>(cb, cb_bf, cnorm);

  for (int q = 0; q < Q_; ++q) {
    const float*  cbq  = cb + (size_t)q * C_ * D_;
    const bf16_t* cbbq = cb_bf + (size_t)q * C_ * D_;
    k_gemm_argmin<<<M_ / MT, 256, 0, stream>>>(res_bf, cbbq, cnorm + (size_t)q * C_, amin);
    k_update<<<M_ / 8, 256, 0, stream>>>(residual, res_bf, cbq, amin, qacc, loss_rows,
                                         q == 0 ? 1 : 0);
  }

  k_reduce_loss<<<1, 256, 0, stream>>>(loss_rows, loss);
  k_finalize<<<tg, tb, 0, stream>>>(qacc, out, loss);
}
